// AFT_FULL_3289944949032
// MI455X (gfx1250) — matrix operand order repaired and checked by reading, not compiled, not run on hardware
//
#include <hip/hip_runtime.h>

typedef __attribute__((ext_vector_type(16))) _Float16 v16h;
typedef __attribute__((ext_vector_type(8)))  _Float16 v8h;
typedef __attribute__((ext_vector_type(4)))  _Float16 v4h;
typedef __attribute__((ext_vector_type(8)))  float    v8f;

#define BSZ   32
#define SEQ   1024
#define DIM   512
#define MTOT  (BSZ * SEQ)     // 32768 rows of x
#define D2    (2 * DIM)       // 1024 (num || den feature dim)

// ---------- WMMA fragment loaders (16-bit, wave32, contiguous-K order) ----------

// A 16x32 f16: lane L holds row (L&15); K chunk base c = 16*(L>>4);
// K = [c..c+15], contiguous. 'p' points at rowbase + c (+k).
static __device__ __forceinline__ v16h load_frag_a(const _Float16* p) {
    const v8h lo = *(const v8h*)(p);
    const v8h hi = *(const v8h*)(p + 8);
    v16h r;
#pragma unroll
    for (int t = 0; t < 8; ++t) { r[t] = lo[t]; r[t + 8] = hi[t]; }
    return r;
}

// B 32x16 f16: lane L holds column (L&15); K = [16*(L>>4) .. +15], contiguous
// when B is stored column-major (i.e. we point into W row n / Yt row d).
static __device__ __forceinline__ v16h load_frag_b(const _Float16* p) {
    const v8h lo = *(const v8h*)(p);
    const v8h hi = *(const v8h*)(p + 8);
    v16h r;
#pragma unroll
    for (int t = 0; t < 8; ++t) { r[t] = lo[t]; r[t + 8] = hi[t]; }
    return r;
}

// ---------- Stage 1: conversions ----------

__global__ void cvt_f16_x4(const float* __restrict__ src, _Float16* __restrict__ dst, int n) {
    int i = (blockIdx.x * blockDim.x + threadIdx.x) * 4;
    if (i + 3 < n) {
        float4 v = *(const float4*)(src + i);
        v4h o;
        o[0] = (_Float16)v.x; o[1] = (_Float16)v.y;
        o[2] = (_Float16)v.z; o[3] = (_Float16)v.w;
        *(v4h*)(dst + i) = o;
    }
}

__global__ void cvt_exp_f16(const float* __restrict__ src, _Float16* __restrict__ dst, int n) {
    int i = blockIdx.x * blockDim.x + threadIdx.x;
    if (i < n) dst[i] = (_Float16)__expf(src[i]);
}

// ---------- Stage 2: fused QKV projection (one wave = one 16x16 tile, 3 accumulators) ----------

__global__ __launch_bounds__(256)
void qkv_wmma(const _Float16* __restrict__ xh,   // [MTOT, DIM]
              const _Float16* __restrict__ wq,   // [DIM, DIM] row-major (== B col-major)
              const _Float16* __restrict__ wk,
              const _Float16* __restrict__ wv,
              const float* __restrict__ bq,
              const float* __restrict__ bk,
              const float* __restrict__ bv,
              _Float16* __restrict__ S,          // [MTOT, DIM] sigmoid(q)
              _Float16* __restrict__ Yt)         // [BSZ, D2, SEQ]: rows 0..511 ek*v, 512..1023 ek
{
    const int lane = threadIdx.x & 31;
    const int wid  = threadIdx.x >> 5;
    const int tile = blockIdx.x * 8 + wid;     // 2048 * 32 = 65536 tiles
    const int tm = tile >> 5;                  // M tile (0..2047)
    const int tn = tile & 31;                  // N tile (0..31)
    const int m0 = tm << 4;
    const int n0 = tn << 4;

    const int lh = lane >> 4;                  // 0 / 1 (lane half)
    const int ll = lane & 15;

    const _Float16* pa  = xh + (size_t)(m0 + ll) * DIM + (lh << 4);
    const int d = n0 + ll;                     // this lane's output feature column
    const _Float16* pbq = wq + (size_t)d * DIM + (lh << 4);
    const _Float16* pbk = wk + (size_t)d * DIM + (lh << 4);
    const _Float16* pbv = wv + (size_t)d * DIM + (lh << 4);

    v8f accq = {}, acck = {}, accv = {};
#pragma unroll 4
    for (int k = 0; k < DIM; k += 32) {
        __builtin_prefetch(pa + k + 128, 0, 1);     // next A chunks -> global_prefetch
        v16h a  = load_frag_a(pa + k);
        v16h b0 = load_frag_b(pbq + k);
        v16h b1 = load_frag_b(pbk + k);
        v16h b2 = load_frag_b(pbv + k);
        accq = __builtin_amdgcn_wmma_f32_16x16x32_f16(false, a, false, b0, (short)0, accq, false, false);
        acck = __builtin_amdgcn_wmma_f32_16x16x32_f16(false, a, false, b1, (short)0, acck, false, false);
        accv = __builtin_amdgcn_wmma_f32_16x16x32_f16(false, a, false, b2, (short)0, accv, false, false);
    }

    const float bqv = bq[d], bkv = bk[d], bvv = bv[d];
    const int b     = m0 >> 10;                       // batch (16-row tiles never cross batches)
    const int jbase = (m0 & (SEQ - 1)) + (lh << 3);   // 8 contiguous tokens per lane

    v8h ekv_h, ek_h;
#pragma unroll
    for (int r = 0; r < 8; ++r) {
        const int row = m0 + (lh << 3) + r;           // global token row
        const float qv = accq[r] + bqv;
        const float sg = 1.0f / (1.0f + __expf(-qv));
        const float ek = __expf(acck[r] + bkv);
        const float vv = accv[r] + bvv;
        ek_h[r]  = (_Float16)ek;
        ekv_h[r] = (_Float16)(ek * vv);
        S[(size_t)row * DIM + d] = (_Float16)sg;
    }
    _Float16* yb = Yt + (size_t)b * D2 * SEQ;
    *(v8h*)(yb + (size_t)d * SEQ + jbase)         = ekv_h;   // numerator operand
    *(v8h*)(yb + (size_t)(DIM + d) * SEQ + jbase) = ek_h;    // denominator operand
}

// ---------- Stage 3: AFT attention GEMM: out2[b] = EB @ [ek*v | ek] ----------
// One wave computes a 16x64 slab (4 accumulators) so the EB A-fragment is
// loaded once per 4 WMMAs.

__global__ __launch_bounds__(256)
void aft_wmma(const _Float16* __restrict__ ebh,   // [SEQ, SEQ] = exp(pos_bias), f16
              const _Float16* __restrict__ Yt,    // [BSZ, D2, SEQ]
              float* __restrict__ out2)           // [BSZ, SEQ, D2]
{
    const int lane = threadIdx.x & 31;
    const int wid  = threadIdx.x >> 5;
    const int tile = blockIdx.x * 8 + wid;     // 32 * 64 * 16 = 32768 slabs
    const int b  = tile >> 10;                 // batch
    const int ti = (tile >> 4) & 63;           // i tile (16 rows)
    const int tq = tile & 15;                  // d quad (64 cols)
    const int i0 = ti << 4;
    const int d0 = tq << 6;

    const int lh = lane >> 4;
    const int ll = lane & 15;

    const _Float16* pa = ebh + (size_t)(i0 + ll) * SEQ + (lh << 4);
    const _Float16* yb = Yt + (size_t)b * D2 * SEQ;
    const _Float16* pb0 = yb + (size_t)(d0 +  0 + ll) * SEQ + (lh << 4);
    const _Float16* pb1 = yb + (size_t)(d0 + 16 + ll) * SEQ + (lh << 4);
    const _Float16* pb2 = yb + (size_t)(d0 + 32 + ll) * SEQ + (lh << 4);
    const _Float16* pb3 = yb + (size_t)(d0 + 48 + ll) * SEQ + (lh << 4);

    v8f a0 = {}, a1 = {}, a2 = {}, a3 = {};
#pragma unroll 2
    for (int j = 0; j < SEQ; j += 32) {
        __builtin_prefetch(pa + j + 64, 0, 1);
        v16h a  = load_frag_a(pa + j);
        v16h q0 = load_frag_b(pb0 + j);
        v16h q1 = load_frag_b(pb1 + j);
        v16h q2 = load_frag_b(pb2 + j);
        v16h q3 = load_frag_b(pb3 + j);
        a0 = __builtin_amdgcn_wmma_f32_16x16x32_f16(false, a, false, q0, (short)0, a0, false, false);
        a1 = __builtin_amdgcn_wmma_f32_16x16x32_f16(false, a, false, q1, (short)0, a1, false, false);
        a2 = __builtin_amdgcn_wmma_f32_16x16x32_f16(false, a, false, q2, (short)0, a2, false, false);
        a3 = __builtin_amdgcn_wmma_f32_16x16x32_f16(false, a, false, q3, (short)0, a3, false, false);
    }

    float* ob = out2 + (size_t)b * SEQ * D2;
#pragma unroll
    for (int r = 0; r < 8; ++r) {
        float* row = ob + (size_t)(i0 + (lh << 3) + r) * D2 + ll;
        row[d0 +  0] = a0[r];
        row[d0 + 16] = a1[r];
        row[d0 + 32] = a2[r];
        row[d0 + 48] = a3[r];
    }
}

// ---------- Stage 4: out = sigmoid(q) * num / den (4 elements / thread) ----------

__global__ void finalize(const _Float16* __restrict__ S,
                         const float* __restrict__ out2,
                         float* __restrict__ out)
{
    const int idx = (blockIdx.x * blockDim.x + threadIdx.x) * 4;  // BSZ*SEQ*DIM elems
    const int d  = idx & (DIM - 1);
    const int bi = idx >> 9;                                       // b*SEQ + i
    const float4 num = *(const float4*)(out2 + (size_t)bi * D2 + d);
    const float4 den = *(const float4*)(out2 + (size_t)bi * D2 + DIM + d);
    const v4h    sg  = *(const v4h*)(S + idx);
    float4 o;
    o.x = (float)sg[0] * (num.x / den.x);
    o.y = (float)sg[1] * (num.y / den.y);
    o.z = (float)sg[2] * (num.z / den.z);
    o.w = (float)sg[3] * (num.w / den.w);
    *(float4*)(out + idx) = o;
}

// ---------- Host launcher ----------

extern "C" void kernel_launch(void* const* d_in, const int* in_sizes, int n_in,
                              void* d_out, int out_size, void* d_ws, size_t ws_size,
                              hipStream_t stream) {
    (void)in_sizes; (void)n_in; (void)out_size; (void)ws_size;
    const float* x  = (const float*)d_in[0];
    const float* Wq = (const float*)d_in[1];
    const float* bq = (const float*)d_in[2];
    const float* Wk = (const float*)d_in[3];
    const float* bk = (const float*)d_in[4];
    const float* Wv = (const float*)d_in[5];
    const float* bv = (const float*)d_in[6];
    const float* pb = (const float*)d_in[7];
    float* out = (float*)d_out;

    char* ws = (char*)d_ws;
    _Float16* xh  = (_Float16*)ws;  ws += (size_t)MTOT * DIM * 2;     // 32 MB
    _Float16* wqh = (_Float16*)ws;  ws += (size_t)DIM * DIM * 2;      // 0.5 MB
    _Float16* wkh = (_Float16*)ws;  ws += (size_t)DIM * DIM * 2;
    _Float16* wvh = (_Float16*)ws;  ws += (size_t)DIM * DIM * 2;
    _Float16* ebh = (_Float16*)ws;  ws += (size_t)SEQ * SEQ * 2;      // 2 MB
    _Float16* S   = (_Float16*)ws;  ws += (size_t)MTOT * DIM * 2;     // 32 MB
    _Float16* Yt  = (_Float16*)ws;  ws += (size_t)BSZ * D2 * SEQ * 2; // 64 MB
    float*    o2  = (float*)ws;     // [BSZ, SEQ, D2] 128 MB

    cvt_f16_x4<<<(MTOT * DIM) / 4 / 256, 256, 0, stream>>>(x, xh, MTOT * DIM);
    cvt_f16_x4<<<(DIM * DIM) / 4 / 256, 256, 0, stream>>>(Wq, wqh, DIM * DIM);
    cvt_f16_x4<<<(DIM * DIM) / 4 / 256, 256, 0, stream>>>(Wk, wkh, DIM * DIM);
    cvt_f16_x4<<<(DIM * DIM) / 4 / 256, 256, 0, stream>>>(Wv, wvh, DIM * DIM);
    cvt_exp_f16<<<(SEQ * SEQ) / 256, 256, 0, stream>>>(pb, ebh, SEQ * SEQ);

    // 65536 16x16 tiles, 8 waves/block
    qkv_wmma<<<8192, 256, 0, stream>>>(xh, wqh, wkh, wvh, bq, bk, bv, S, Yt);
    // 32768 16x64 slabs, 8 waves/block
    aft_wmma<<<4096, 256, 0, stream>>>(ebh, Yt, o2);
    // 4 elements per thread
    finalize<<<(MTOT * DIM) / 4 / 256, 256, 0, stream>>>(S, o2, out);
}
